// GroupedQueryAttention_33732673143274
// MI455X (gfx1250) — matrix operand order repaired and checked by reading, not compiled, not run on hardware
//
#include <hip/hip_runtime.h>

// ---------------------------------------------------------------------------
// Grouped (block-diagonal) multi-head attention for MI455X / gfx1250.
// B=4, N=8192, E=1024, H=16, D=64, G=64, GS=128.
// All matmuls: v_wmma_f32_16x16x32_bf16 (bf16 in, f32 accumulate).
// Weights pre-transposed+converted to bf16 once (amortized over 256 M-tiles)
// so GEMM LDS staging is pure coalesced copies (no in-tile transposes, no
// LDS bank serialization).  GEMM K-loops are double-buffered to overlap HBM
// loads with WMMA issue.
// ---------------------------------------------------------------------------

typedef unsigned short u16;
typedef unsigned int   u32;
typedef __attribute__((ext_vector_type(16))) __bf16 v16bf;
typedef __attribute__((ext_vector_type(8)))  __bf16 v8bf;
typedef __attribute__((ext_vector_type(8)))  float  v8f;

#define CB 4
#define CN 8192
#define CE 1024
#define CH 16
#define CD 64
#define CG 64
#define CGS 128

__device__ __forceinline__ u16 f2bf(float f) {
  u32 u = __builtin_bit_cast(u32, f);
  u32 r = u + 0x7fffu + ((u >> 16) & 1u);   // round-to-nearest-even
  return (u16)(r >> 16);
}

// A fragment (16x32, bf16). Lane m in [0,16): row m; low lane half holds
// K[0..16), high half holds K[16..32) contiguously.
__device__ __forceinline__ v16bf load_a_frag(const u16* base, int stride,
                                             int r0, int k0, int lane) {
  int m = lane & 15, hf = lane >> 4;
  const u16* p = base + (r0 + m) * stride + k0 + hf * 16;
  union { v16bf v; v8bf h[2]; } u;
  u.h[0] = *(const v8bf*)(p);
  u.h[1] = *(const v8bf*)(p + 8);
  return u.v;
}

// B fragment (32x16, bf16) from an N-major LDS tile (tile[n][k]): lane n holds
// column n; low/high lane half holds K[0..16)/K[16..32) contiguously.
__device__ __forceinline__ v16bf load_b_frag(const u16* base, int stride,
                                             int n0, int k0, int lane) {
  int n = lane & 15, hf = lane >> 4;
  const u16* p = base + (n0 + n) * stride + k0 + hf * 16;
  return *(const v16bf*)(p);
}

__device__ __forceinline__ v8f wmma_bf16(v16bf a, v16bf b, v8f c) {
  return __builtin_amdgcn_wmma_f32_16x16x32_bf16(false, a, false, b,
                                                 (short)0, c, false, false);
}

// ---------------------------------------------------------------------------
// Kernel 0: pre-transpose + bf16-convert the four weight matrices.
// W fp32 [k][n]  ->  Wt bf16 [mat][n][k].   64x64 tiles via padded LDS.
// ---------------------------------------------------------------------------
__global__ __launch_bounds__(256) void transpose_w_kernel(
    const float* __restrict__ Wq, const float* __restrict__ Wk,
    const float* __restrict__ Wv, const float* __restrict__ Wo,
    u16* __restrict__ Wt) {
  __shared__ u16 Ts[64][68];                   // padded: stride 68 avoids bank hits
  const int which = blockIdx.z;
  const float* W = (which == 0) ? Wq : (which == 1) ? Wk
                  : (which == 2) ? Wv : Wo;
  u16* out = Wt + (size_t)which * CE * CE;
  const int k0 = blockIdx.x * 64, n0 = blockIdx.y * 64;
  const int t = threadIdx.x;

#pragma unroll
  for (int i = 0; i < 4; ++i) {
    int c = i * 256 + t;                       // chunk of 4 floats, 16/row
    int row = c >> 4, col = (c & 15) * 4;
    float4 f = *(const float4*)(W + (size_t)(k0 + row) * CE + n0 + col);
    Ts[row][col + 0] = f2bf(f.x);
    Ts[row][col + 1] = f2bf(f.y);
    Ts[row][col + 2] = f2bf(f.z);
    Ts[row][col + 3] = f2bf(f.w);
  }
  __syncthreads();
#pragma unroll
  for (int i = 0; i < 2; ++i) {
    int c = i * 256 + t;                       // 512 chunks of 8 k-values
    int n = c >> 3, kc = (c & 7) * 8;
    u16 tmp[8];
#pragma unroll
    for (int j = 0; j < 8; ++j) tmp[j] = Ts[kc + j][n];
    *(uint4*)(out + (size_t)(n0 + n) * CE + k0 + kc) = *(uint4*)tmp;
  }
}

// ---------------------------------------------------------------------------
// Kernel 1: fused QKV projection.  Y = X @ W + bias.
// X fp32 [32768,1024]; W from pre-transposed bf16 Wt [n][k].
// Q,K outputs -> bf16 [B,H,N,D].  V output -> bf16 [B,H,G][D][GS] (d-major,
// so the attention kernel needs no V transpose).
// Double-buffered LDS K-loop (32 iters of K=32).
// ---------------------------------------------------------------------------
__global__ __launch_bounds__(256) void proj_qkv_kernel(
    const float* __restrict__ q, const float* __restrict__ k,
    const float* __restrict__ v, const u16* __restrict__ Wt,
    const float* __restrict__ bq, const float* __restrict__ bk,
    const float* __restrict__ bv,
    u16* __restrict__ qp, u16* __restrict__ kp, u16* __restrict__ vtp) {
  __shared__ alignas(64) u16 As[2][128][32];   // X tile, row-major (M x K)
  __shared__ alignas(64) u16 Bs[2][128][32];   // W tile, N-major   (N x K)

  const int which = blockIdx.z;
  const float* X    = (which == 0) ? q  : (which == 1) ? k  : v;
  const float* bias = (which == 0) ? bq : (which == 1) ? bk : bv;
  u16*         Y    = (which == 0) ? qp : (which == 1) ? kp : vtp;
  const u16*   Wb   = Wt + (size_t)which * CE * CE;

  const int n0 = blockIdx.x * 128;
  const int m0 = blockIdx.y * 128;
  const int t = threadIdx.x, lane = t & 31, w = t >> 5;
  const int wm = w >> 1, wn = w & 1;           // 4x2 wave grid

  v8f zero = {};
  v8f acc[8];
#pragma unroll
  for (int i = 0; i < 8; ++i) acc[i] = zero;

  float4 aF[4];                                // in-flight next A tile (fp32)
  uint4  bF[2];                                // in-flight next B tile (bf16)

  auto load_tile = [&](int it) {
    int kk = it * 32;
#pragma unroll
    for (int i = 0; i < 4; ++i) {              // A: 1024 chunks of 4 floats
      int c = i * 256 + t, row = c >> 3, col = (c & 7) * 4;
      aF[i] = *(const float4*)(X + (size_t)(m0 + row) * CE + kk + col);
    }
#pragma unroll
    for (int i = 0; i < 2; ++i) {              // B: 512 chunks of 8 bf16
      int c = i * 256 + t, row = c >> 2, col = (c & 3) * 8;
      bF[i] = *(const uint4*)(Wb + (size_t)(n0 + row) * CE + kk + col);
    }
  };
  auto store_tile = [&](int buf) {
#pragma unroll
    for (int i = 0; i < 4; ++i) {
      int c = i * 256 + t, row = c >> 3, col = (c & 7) * 4;
      u32 lo = (u32)f2bf(aF[i].x) | ((u32)f2bf(aF[i].y) << 16);
      u32 hi = (u32)f2bf(aF[i].z) | ((u32)f2bf(aF[i].w) << 16);
      *(u32*)&As[buf][row][col]     = lo;
      *(u32*)&As[buf][row][col + 2] = hi;
    }
#pragma unroll
    for (int i = 0; i < 2; ++i) {
      int c = i * 256 + t, row = c >> 2, col = (c & 3) * 8;
      *(uint4*)&Bs[buf][row][col] = bF[i];
    }
  };

  load_tile(0);
  store_tile(0);
  __syncthreads();

  for (int it = 0; it < 32; ++it) {
    const int cur = it & 1;
    if (it + 1 < 32) load_tile(it + 1);        // HBM loads in flight ...

    v16bf af[2], bfr[4];
#pragma unroll
    for (int sm = 0; sm < 2; ++sm)
      af[sm] = load_a_frag(&As[cur][0][0], 32, wm * 32 + sm * 16, 0, lane);
#pragma unroll
    for (int sn = 0; sn < 4; ++sn)
      bfr[sn] = load_b_frag(&Bs[cur][0][0], 32, wn * 64 + sn * 16, 0, lane);
#pragma unroll
    for (int sm = 0; sm < 2; ++sm)
#pragma unroll
      for (int sn = 0; sn < 4; ++sn)           // ... while WMMAs issue
        acc[sm * 4 + sn] = wmma_bf16(af[sm], bfr[sn], acc[sm * 4 + sn]);

    if (it + 1 < 32) store_tile(cur ^ 1);
    __syncthreads();
  }

  // Epilogue: +bias, write bf16.
  const int hf = lane >> 4, nl = lane & 15;
#pragma unroll
  for (int sm = 0; sm < 2; ++sm) {
#pragma unroll
    for (int sn = 0; sn < 4; ++sn) {
      v8f c = acc[sm * 4 + sn];
      int cg = n0 + wn * 64 + sn * 16 + nl;    // global output column
      float bi = bias[cg];
      int h = cg >> 6, d = cg & 63;
#pragma unroll
      for (int r = 0; r < 8; ++r) {
        int mrow = m0 + wm * 32 + sm * 16 + hf * 8 + r;
        int b = mrow >> 13, n = mrow & (CN - 1);
        u16 y = f2bf(c[r] + bi);
        if (which != 2) {
          Y[(((size_t)b * CH + h) * CN + n) * CD + d] = y;
        } else {                               // V: d-major group layout
          int g = n >> 7, rl = n & (CGS - 1);
          Y[((((size_t)b * CH + h) * CG + g) * CD + d) * CGS + rl] = y;
        }
      }
    }
  }
}

// ---------------------------------------------------------------------------
// Kernel 2: block-diagonal attention for one (b,h,g).
// S = Q K^T * scale (128x128), softmax rows, O = P V (128x64).
// 256 threads = 8 waves; wave w owns rows [w*16, w*16+16).
// LDS: Qs(16K) Ks(16K) Vt(16K); Ps(32K) overlays Qs+Ks.  48 KB total.
// ---------------------------------------------------------------------------
__global__ __launch_bounds__(256) void attn_kernel(
    const u16* __restrict__ qp, const u16* __restrict__ kp,
    const u16* __restrict__ vtp, u16* __restrict__ op,
    float* __restrict__ attnw) {
  __shared__ alignas(64) u16 smem[128 * 64 + 128 * 64 + 64 * 128];
  u16* Qs = smem;                 // [128][64]
  u16* Ks = smem + 128 * 64;      // [128][64]  (natural layout == N-major B)
  u16* Vt = smem + 2 * 128 * 64;  // [64][128]  (already d-major in memory)
  u16* Ps = smem;                 // [128][128] overlays Qs+Ks

  const int g = blockIdx.x, h = blockIdx.y, b = blockIdx.z;
  const size_t qkbase = (((size_t)b * CH + h) * CN + (size_t)g * CGS) * CD;
  const u16* vbase = vtp + (((size_t)b * CH + h) * CG + g) * CD * CGS;
  const int t = threadIdx.x, lane = t & 31, w = t >> 5;

  // Stage Q, K (row-major 128x64) and V (d-major 64x128): straight copies.
#pragma unroll
  for (int i = 0; i < 4; ++i) {
    int c = i * 256 + t;                       // 1024 chunks of 8 bf16
    int row = c >> 3, col = (c & 7) * 8;
    *(uint4*)&Qs[row * 64 + col] =
        *(const uint4*)(qp + qkbase + row * 64 + col);
    *(uint4*)&Ks[row * 64 + col] =
        *(const uint4*)(kp + qkbase + row * 64 + col);
    int vrow = c >> 4, vcol = (c & 15) * 8;    // Vt is 64 x 128
    *(uint4*)&Vt[vrow * 128 + vcol] =
        *(const uint4*)(vbase + vrow * 128 + vcol);
  }
  __syncthreads();

  // S = Q K^T over this wave's 16-row strip (8 column subtiles).
  v8f zero = {};
  v8f acc[8];
#pragma unroll
  for (int i = 0; i < 8; ++i) acc[i] = zero;
#pragma unroll
  for (int k0 = 0; k0 < 64; k0 += 32) {
    v16bf af = load_a_frag(Qs, 64, w * 16, k0, lane);
#pragma unroll
    for (int sn = 0; sn < 8; ++sn) {
      v16bf bf_ = load_b_frag(Ks, 64, sn * 16, k0, lane);
      acc[sn] = wmma_bf16(af, bf_, acc[sn]);
    }
  }

  // Row softmax. C layout: VGPR r <-> row r (lanes 0-15) / r+8 (lanes 16-31);
  // xor masks 1,2,4,8 stay inside each 16-lane half.
  const float scale = 0.125f;                  // 1/sqrt(64)
  float rsum[8];
#pragma unroll
  for (int r = 0; r < 8; ++r) {
    float m = -3.0e38f;
#pragma unroll
    for (int s = 0; s < 8; ++s) {
      float x = acc[s][r] * scale;
      acc[s][r] = x;
      m = fmaxf(m, x);
    }
#pragma unroll
    for (int off = 1; off < 16; off <<= 1) m = fmaxf(m, __shfl_xor(m, off, 32));
    float ssum = 0.f;
#pragma unroll
    for (int s = 0; s < 8; ++s) {
      float e = __expf(acc[s][r] - m);
      acc[s][r] = e;
      ssum += e;
    }
#pragma unroll
    for (int off = 1; off < 16; off <<= 1) ssum += __shfl_xor(ssum, off, 32);
    rsum[r] = ssum;
  }

  __syncthreads();                             // Qs/Ks reads done before Ps writes

  const int hf = lane >> 4, nl = lane & 15;
  const bool last = (g == CG - 1);
#pragma unroll
  for (int s = 0; s < 8; ++s) {
    int col = s * 16 + nl;
#pragma unroll
    for (int r = 0; r < 8; ++r) {
      float p = acc[s][r] / rsum[r];
      int row = w * 16 + hf * 8 + r;
      Ps[row * 128 + col] = f2bf(p);
      if (last)
        attnw[(((size_t)b * CH + h) * CGS + row) * CGS + col] = p;
    }
  }
  __syncthreads();

  // O = P @ V : 128x64; this wave: 16 rows x 4 column subtiles.
  v8f oacc[4];
#pragma unroll
  for (int i = 0; i < 4; ++i) oacc[i] = zero;
#pragma unroll
  for (int k0 = 0; k0 < 128; k0 += 32) {
    v16bf af = load_a_frag(Ps, 128, w * 16, k0, lane);
#pragma unroll
    for (int sn = 0; sn < 4; ++sn) {
      v16bf bf_ = load_b_frag(Vt, 128, sn * 16, k0, lane);
      oacc[sn] = wmma_bf16(af, bf_, oacc[sn]);
    }
  }

  // Write O (bf16) into [B,N,E] with e = h*64 + d (ready for out-projection).
#pragma unroll
  for (int sn = 0; sn < 4; ++sn) {
    int d = sn * 16 + nl;
#pragma unroll
    for (int r = 0; r < 8; ++r) {
      int row = w * 16 + hf * 8 + r;
      size_t nglob = (size_t)g * CGS + row;
      op[((size_t)b * CN + nglob) * CE + h * CD + d] = f2bf(oacc[sn][r]);
    }
  }
}

// ---------------------------------------------------------------------------
// Kernel 3: output projection.  out = op(bf16 [32768,1024]) @ Wo + bo, fp32.
// Double-buffered like kernel 1; both A and B tiles are bf16 copies.
// ---------------------------------------------------------------------------
__global__ __launch_bounds__(256) void out_proj_kernel(
    const u16* __restrict__ Xb, const u16* __restrict__ Wbt,
    const float* __restrict__ bias, float* __restrict__ out) {
  __shared__ alignas(64) u16 As[2][128][32];
  __shared__ alignas(64) u16 Bs[2][128][32];

  const int n0 = blockIdx.x * 128;
  const int m0 = blockIdx.y * 128;
  const int t = threadIdx.x, lane = t & 31, w = t >> 5;
  const int wm = w >> 1, wn = w & 1;

  v8f zero = {};
  v8f acc[8];
#pragma unroll
  for (int i = 0; i < 8; ++i) acc[i] = zero;

  uint4 aF[2], bF[2];
  auto load_tile = [&](int it) {
    int kk = it * 32;
#pragma unroll
    for (int i = 0; i < 2; ++i) {
      int c = i * 256 + t, row = c >> 2, col = (c & 3) * 8;
      aF[i] = *(const uint4*)(Xb  + (size_t)(m0 + row) * CE + kk + col);
      bF[i] = *(const uint4*)(Wbt + (size_t)(n0 + row) * CE + kk + col);
    }
  };
  auto store_tile = [&](int buf) {
#pragma unroll
    for (int i = 0; i < 2; ++i) {
      int c = i * 256 + t, row = c >> 2, col = (c & 3) * 8;
      *(uint4*)&As[buf][row][col] = aF[i];
      *(uint4*)&Bs[buf][row][col] = bF[i];
    }
  };

  load_tile(0);
  store_tile(0);
  __syncthreads();

  for (int it = 0; it < 32; ++it) {
    const int cur = it & 1;
    if (it + 1 < 32) load_tile(it + 1);

    v16bf af[2], bfr[4];
#pragma unroll
    for (int sm = 0; sm < 2; ++sm)
      af[sm] = load_a_frag(&As[cur][0][0], 32, wm * 32 + sm * 16, 0, lane);
#pragma unroll
    for (int sn = 0; sn < 4; ++sn)
      bfr[sn] = load_b_frag(&Bs[cur][0][0], 32, wn * 64 + sn * 16, 0, lane);
#pragma unroll
    for (int sm = 0; sm < 2; ++sm)
#pragma unroll
      for (int sn = 0; sn < 4; ++sn)
        acc[sm * 4 + sn] = wmma_bf16(af[sm], bfr[sn], acc[sm * 4 + sn]);

    if (it + 1 < 32) store_tile(cur ^ 1);
    __syncthreads();
  }

  const int hf = lane >> 4, nl = lane & 15;
#pragma unroll
  for (int sm = 0; sm < 2; ++sm) {
#pragma unroll
    for (int sn = 0; sn < 4; ++sn) {
      v8f c = acc[sm * 4 + sn];
      int cg = n0 + wn * 64 + sn * 16 + nl;
      float bi = bias[cg];
#pragma unroll
      for (int r = 0; r < 8; ++r) {
        int mrow = m0 + wm * 32 + sm * 16 + hf * 8 + r;
        out[(size_t)mrow * CE + cg] = c[r] + bi;
      }
    }
  }
}

// ---------------------------------------------------------------------------
extern "C" void kernel_launch(void* const* d_in, const int* in_sizes, int n_in,
                              void* d_out, int out_size, void* d_ws,
                              size_t ws_size, hipStream_t stream) {
  const float* q  = (const float*)d_in[0];
  const float* k  = (const float*)d_in[1];
  const float* v  = (const float*)d_in[2];
  const float* Wq = (const float*)d_in[3];
  const float* bq = (const float*)d_in[4];
  const float* Wk = (const float*)d_in[5];
  const float* bk = (const float*)d_in[6];
  const float* Wv = (const float*)d_in[7];
  const float* bv = (const float*)d_in[8];
  const float* Wo = (const float*)d_in[9];
  const float* bo = (const float*)d_in[10];
  float* out = (float*)d_out;

  const size_t wsz = (size_t)CE * CE;             // 1M elems per weight matrix
  const size_t per = (size_t)CB * CH * CN * CD;   // 33.5M bf16 elems each
  u16* Wt  = (u16*)d_ws;                          // [4][1024][1024] bf16, [n][k]
  u16* qp  = Wt + 4 * wsz;                        // [B,H,N,D]
  u16* kp  = qp + per;                            // [B,H,N,D]
  u16* vtp = kp + per;                            // [B,H,G][D][GS]
  u16* op  = vtp + per;                           // [B,N,E] attention output
  float* attnw = out + (size_t)CB * CN * CE;      // second output region

  // 0) One-time weight transpose + bf16 convert (amortized over 256 M-tiles).
  transpose_w_kernel<<<dim3(CE / 64, CE / 64, 4), 256, 0, stream>>>(
      Wq, Wk, Wv, Wo, Wt);

  // 1) QKV projections: 3 x (256 x 8) tiles of 128x128.
  proj_qkv_kernel<<<dim3(CE / 128, (CB * CN) / 128, 3), 256, 0, stream>>>(
      q, k, v, Wt, bq, bk, bv, qp, kp, vtp);

  // 2) Block-diagonal attention: one block per (g, h, b).
  attn_kernel<<<dim3(CG, CH, CB), 256, 0, stream>>>(qp, kp, vtp, op, attnw);

  // 3) Output projection -> fp32 d_out.
  out_proj_kernel<<<dim3(CE / 128, (CB * CN) / 128), 256, 0, stream>>>(
      op, Wt + 3 * wsz, bo, out);
}
